// GCN_11321533792312
// MI455X (gfx1250) — compile-verified, matrix operand order checked by reading, not run on hardware
//
#include <hip/hip_runtime.h>
#include <hip/hip_bf16.h>

typedef __attribute__((ext_vector_type(16))) __bf16 v16bf;
typedef __attribute__((ext_vector_type(8)))  __bf16 v8bf;
typedef __attribute__((ext_vector_type(8)))  float  v8f;

#define HID 128
#define IN_DIM 100

// ---------------- degree / normalization ----------------
__global__ void gcn_deg_edges(const int* __restrict__ dst, float* __restrict__ deg, int E) {
    int e = blockIdx.x * blockDim.x + threadIdx.x;
    if (e < E) atomicAdd(&deg[dst[e]], 1.0f);
}

__global__ void gcn_make_dinv(float* __restrict__ deg_dinv, int n) {
    int i = blockIdx.x * blockDim.x + threadIdx.x;
    if (i < n) {
        float d = deg_dinv[i] + 1.0f;   // +1 self loop; always > 0
        deg_dinv[i] = rsqrtf(d);
    }
}

// ---------------- conversions ----------------
// x [n, IN_DIM] f32 -> xb [n, HID] bf16, zero-padded K
__global__ void gcn_cvt_x(const float* __restrict__ x, __bf16* __restrict__ xb, int n) {
    int idx = blockIdx.x * blockDim.x + threadIdx.x;
    if (idx >= n * HID) return;
    int i = idx >> 7, k = idx & 127;
    float v = (k < IN_DIM) ? x[(size_t)i * IN_DIM + k] : 0.0f;
    xb[idx] = (__bf16)v;
}

// W [Krows, HID] row-major f32 -> WT [HID, HID] bf16 (transposed, K zero-padded)
__global__ void gcn_cvt_w(const float* __restrict__ W, __bf16* __restrict__ WT, int Krows) {
    int idx = blockIdx.x * blockDim.x + threadIdx.x;   // HID*HID threads
    if (idx >= HID * HID) return;
    int ncol = idx >> 7, k = idx & 127;
    float v = (k < Krows) ? W[(size_t)k * HID + ncol] : 0.0f;
    WT[(size_t)ncol * HID + k] = (__bf16)v;
}

// ---------------- WMMA GEMM: C[n,128] = A[n,128](bf16) @ WT^T ----------------
// One wave computes a 32(M) x 128(N) stripe: 2 M-tiles so every B-fragment
// (ds_load_b128 pair) feeds two v_wmma ops.
__global__ __launch_bounds__(256) void gcn_gemm_wmma(const __bf16* __restrict__ A,
                                                     const __bf16* __restrict__ WT,
                                                     float* __restrict__ C, int n) {
    __shared__ __bf16 sW[HID * HID];           // 32 KB, B matrix (transposed) in LDS
    {
        const uint4* s = (const uint4*)WT;     // 2048 x 16B
        uint4* d = (uint4*)sW;
        for (int i = threadIdx.x; i < 2048; i += 256) d[i] = s[i];
    }
    __syncthreads();

    const int wave  = threadIdx.x >> 5;
    const int lane  = threadIdx.x & 31;
    const int mPair = blockIdx.x * 8 + wave;   // 32 rows per wave
    const int mBase = mPair * 32;
    if (mBase >= n) return;                    // wave-uniform guard (EXEC stays all-1 for WMMA)

    const int half   = (lane < 16) ? 0 : 8;    // K sub-chunk per ISA 16-bit A layout
    const int colSel = lane & 15;
    int rowA0 = mBase + (lane & 15);
    int rowA1 = rowA0 + 16;
    if (rowA0 >= n) rowA0 = n - 1;             // clamped reads; stores are guarded
    if (rowA1 >= n) rowA1 = n - 1;

    v8f acc0[8], acc1[8];
#pragma unroll
    for (int nt = 0; nt < 8; ++nt) {
        acc0[nt] = (v8f){0,0,0,0,0,0,0,0};
        acc1[nt] = (v8f){0,0,0,0,0,0,0,0};
    }

#pragma unroll
    for (int kk = 0; kk < 4; ++kk) {           // K = 4 x 32
        const int kb = kk * 32 + half;

        // A fragments for both M-tiles (two 16B global loads each)
        const __bf16* ap0 = A + (size_t)rowA0 * HID + kb;
        const __bf16* ap1 = A + (size_t)rowA1 * HID + kb;
        v8bf a0lo = *(const v8bf*)ap0;
        v8bf a0hi = *(const v8bf*)(ap0 + 16);
        v8bf a1lo = *(const v8bf*)ap1;
        v8bf a1hi = *(const v8bf*)(ap1 + 16);
        v16bf af0, af1;
#pragma unroll
        for (int i = 0; i < 8; ++i) {
            af0[i] = a0lo[i]; af0[i + 8] = a0hi[i];
            af1[i] = a1lo[i]; af1[i + 8] = a1hi[i];
        }

        // Preload all 8 B-fragments for this k-step, then burst 16 WMMAs
        v16bf bfrag[8];
#pragma unroll
        for (int nt = 0; nt < 8; ++nt) {
            const __bf16* bp = sW + (size_t)(nt * 16 + colSel) * HID + kb;
            v8bf blo = *(const v8bf*)bp;
            v8bf bhi = *(const v8bf*)(bp + 16);
#pragma unroll
            for (int i = 0; i < 8; ++i) { bfrag[nt][i] = blo[i]; bfrag[nt][i + 8] = bhi[i]; }
        }

#pragma unroll
        for (int nt = 0; nt < 8; ++nt) {
            acc0[nt] = __builtin_amdgcn_wmma_f32_16x16x32_bf16(
                false, af0, false, bfrag[nt], (short)0, acc0[nt], false, false);
            acc1[nt] = __builtin_amdgcn_wmma_f32_16x16x32_bf16(
                false, af1, false, bfrag[nt], (short)0, acc1[nt], false, false);
        }
    }

    // C/D layout: VGPR v -> row base + v + (lane<16 ? 0 : 8), col = nt*16 + (lane&15)
    const int rb0 = mBase + half;
    const int rb1 = mBase + 16 + half;
#pragma unroll
    for (int nt = 0; nt < 8; ++nt) {
#pragma unroll
        for (int v = 0; v < 8; ++v) {
            int r0 = rb0 + v;
            int r1 = rb1 + v;
            if (r0 < n) C[(size_t)r0 * HID + nt * 16 + colSel] = acc0[nt][v];
            if (r1 < n) C[(size_t)r1 * HID + nt * 16 + colSel] = acc1[nt][v];
        }
    }
}

// ---------------- edge scatter-add: agg[dst] += t[src] * dinv[src]*dinv[dst] ----------------
__global__ void gcn_scatter(const float* __restrict__ t, const float* __restrict__ dinv,
                            const int* __restrict__ src, const int* __restrict__ dst,
                            float* __restrict__ agg, int E) {
    long long gid = (long long)blockIdx.x * blockDim.x + threadIdx.x;
    int e = (int)(gid >> 5);
    int lane = (int)(gid & 31);
    if (e >= E) return;
    int s = src[e], d = dst[e];
    float nrm = dinv[s] * dinv[d];
    const float* tp = t + (size_t)s * HID;
    float* ap = agg + (size_t)d * HID;
#pragma unroll
    for (int k = 0; k < 4; ++k) {
        int f = lane + 32 * k;
        atomicAdd(&ap[f], tp[f] * nrm);
    }
}

// ---------------- epilogue: relu(agg + t*dinv^2 + b) ----------------
__global__ void gcn_post_bf16(const float* __restrict__ agg, const float* __restrict__ t,
                              const float* __restrict__ dinv, const float* __restrict__ bias,
                              __bf16* __restrict__ out, int n) {
    int idx = blockIdx.x * blockDim.x + threadIdx.x;
    if (idx >= n * HID) return;
    int i = idx >> 7, f = idx & 127;
    float di = dinv[i];
    float v = agg[idx] + t[idx] * di * di + bias[f];
    out[idx] = (__bf16)fmaxf(v, 0.0f);
}

__global__ void gcn_post_f32(const float* __restrict__ agg, const float* __restrict__ t,
                             const float* __restrict__ dinv, const float* __restrict__ bias,
                             float* __restrict__ out, int n) {
    int idx = blockIdx.x * blockDim.x + threadIdx.x;
    if (idx >= n * HID) return;
    int i = idx >> 7, f = idx & 127;
    float di = dinv[i];
    float v = agg[idx] + t[idx] * di * di + bias[f];
    out[idx] = fmaxf(v, 0.0f);
}

// ---------------- mean pool (atomic accumulate) ----------------
__global__ void gcn_pool(const float* __restrict__ h, const int* __restrict__ batch,
                         float* __restrict__ pooled, float* __restrict__ cnt, int n) {
    long long gid = (long long)blockIdx.x * blockDim.x + threadIdx.x;
    int i = (int)(gid >> 5);
    int lane = (int)(gid & 31);
    if (i >= n) return;
    int g = batch[i];
    const float* hp = h + (size_t)i * HID;
    float* pp = pooled + (size_t)g * HID;
#pragma unroll
    for (int k = 0; k < 4; ++k) {
        int f = lane + 32 * k;
        atomicAdd(&pp[f], hp[f]);
    }
    if (lane == 0) atomicAdd(&cnt[g], 1.0f);
}

// ---------------- FC head: out[g,o] = (pooled[g]/cnt[g]) . Wfc[:,o] + bfc[o] ----------------
__global__ void gcn_fc(const float* __restrict__ pooled, const float* __restrict__ cnt,
                       const float* __restrict__ Wfc, const float* __restrict__ bfc,
                       float* __restrict__ out, int NG, int OUT) {
    int idx = blockIdx.x * blockDim.x + threadIdx.x;
    if (idx >= NG * OUT) return;
    int g = idx / OUT, o = idx % OUT;
    const float* pg = pooled + (size_t)g * HID;
    float s = 0.0f;
#pragma unroll 4
    for (int k = 0; k < HID; ++k) s += pg[k] * Wfc[(size_t)k * OUT + o];
    out[idx] = s / fmaxf(cnt[g], 1.0f) + bfc[o];
}

static inline size_t alignup(size_t v) { return (v + 255) & ~(size_t)255; }

extern "C" void kernel_launch(void* const* d_in, const int* in_sizes, int n_in,
                              void* d_out, int out_size, void* d_ws, size_t ws_size,
                              hipStream_t stream) {
    const float* x    = (const float*)d_in[0];
    const int*   ei   = (const int*)d_in[1];
    const int*   batch= (const int*)d_in[2];
    const float* W1   = (const float*)d_in[4];
    const float* b1   = (const float*)d_in[5];
    const float* W2   = (const float*)d_in[6];
    const float* b2   = (const float*)d_in[7];
    const float* Wfc  = (const float*)d_in[8];
    const float* bfc  = (const float*)d_in[9];
    float* out = (float*)d_out;

    const int N  = in_sizes[2];          // 100000 nodes
    const int E  = in_sizes[1] / 2;      // 1.6M edges
    const int OUT = in_sizes[9];         // 2
    const int NG  = out_size / OUT;      // 1000 graphs
    const int* src = ei;
    const int* dst = ei + E;

    // ---- workspace carve-out ----
    char* p = (char*)d_ws;
    float* dinv   = (float*)p;                 p += alignup((size_t)N * 4);
    float* tbuf   = (float*)p;                 p += alignup((size_t)N * HID * 4);
    float* agg    = (float*)p;                 p += alignup((size_t)N * HID * 4);
    __bf16* hb    = (__bf16*)p;                p += alignup((size_t)N * HID * 2);
    __bf16* WT    = (__bf16*)p;                p += alignup((size_t)HID * HID * 2);
    float* pooled = (float*)p;                 p += alignup((size_t)NG * HID * 4);
    float* cnt    = (float*)p;                 p += alignup((size_t)NG * 4);
    (void)ws_size; (void)n_in;

    const int T = 256;
    const int gN    = (N + T - 1) / T;
    const int gNH   = (N * HID + T - 1) / T;
    const int gE    = (E + T - 1) / T;
    const int gEw   = (int)(((long long)E * 32 + T - 1) / T);
    const int gNw   = (int)(((long long)N * 32 + T - 1) / T);
    const int gGemm = ((N + 31) / 32 + 7) / 8;     // 32 rows per wave, 8 waves per block

    // ---- degrees & normalization ----
    hipMemsetAsync(dinv, 0, (size_t)N * 4, stream);
    gcn_deg_edges<<<gE, T, 0, stream>>>(dst, dinv, E);
    gcn_make_dinv<<<gN, T, 0, stream>>>(dinv, N);

    // ---- layer 1: GEMM (bf16 WMMA) -> scatter -> relu/bias -> bf16 ----
    gcn_cvt_x<<<gNH, T, 0, stream>>>(x, hb, N);
    gcn_cvt_w<<<(HID * HID + T - 1) / T, T, 0, stream>>>(W1, WT, IN_DIM);
    gcn_gemm_wmma<<<gGemm, T, 0, stream>>>(hb, WT, tbuf, N);
    hipMemsetAsync(agg, 0, (size_t)N * HID * 4, stream);
    gcn_scatter<<<gEw, T, 0, stream>>>(tbuf, dinv, src, dst, agg, E);
    gcn_post_bf16<<<gNH, T, 0, stream>>>(agg, tbuf, dinv, b1, hb, N);

    // ---- layer 2 ----
    gcn_cvt_w<<<(HID * HID + T - 1) / T, T, 0, stream>>>(W2, WT, HID);
    gcn_gemm_wmma<<<gGemm, T, 0, stream>>>(hb, WT, tbuf, N);
    hipMemsetAsync(agg, 0, (size_t)N * HID * 4, stream);
    gcn_scatter<<<gEw, T, 0, stream>>>(tbuf, dinv, src, dst, agg, E);
    gcn_post_f32<<<gNH, T, 0, stream>>>(agg, tbuf, dinv, b2, tbuf, N);   // h2 in tbuf (f32)

    // ---- mean pool + FC head ----
    hipMemsetAsync(pooled, 0, (size_t)NG * HID * 4, stream);
    hipMemsetAsync(cnt, 0, (size_t)NG * 4, stream);
    gcn_pool<<<gNw, T, 0, stream>>>(tbuf, batch, pooled, cnt, N);
    gcn_fc<<<(NG * OUT + T - 1) / T, T, 0, stream>>>(pooled, cnt, Wfc, bfc, out, NG, OUT);
}
